// DeepSeekBlock_46935402611128
// MI455X (gfx1250) — matrix operand order repaired and checked by reading, not compiled, not run on hardware
//
#include <hip/hip_runtime.h>
#include <hip/hip_bf16.h>
#include <math.h>

// MoE block for MI455X (gfx1250): bf16 WMMA expert GEMMs, wave32,
// double-buffered LDS, async global->LDS staging, v_fma_mix bf16 converts.
#define NTOK   4096
#define DMODEL 1024
#define DFF    4096
#define NEXPERT 8
#define CAP    2048
#define LN_EPS 1e-5f

#define APITCH 40            // LDS row pitch in halves (80B, 16B-aligned, conflict-free frags)
#define TBUF   (128 * APITCH)   // 128-row tile buffer (halves)
#define T64BUF (64 * APITCH)    // 64-row tile buffer (halves)

typedef __bf16 bf16x16 __attribute__((ext_vector_type(16)));
typedef float  f32x8   __attribute__((ext_vector_type(8)));

struct __align__(16) U4 { unsigned int x, y, z, w; };
struct __align__(16) F4 { float x, y, z, w; };

union Frag { U4 q[2]; bf16x16 v; };

// f32 -> bf16 (RNE) into low 16 bits, via CDNA5 V_FMA_MIXLO_BF16 (VOP3P op 62)
__device__ __forceinline__ unsigned short f2bf(float f) {
  unsigned int r;
  asm("v_fma_mixlo_bf16 %0, %1, 1.0, 0" : "=v"(r) : "v"(f));
  return (unsigned short)r;
}

// pack two floats into one 32-bit word of bf16 (lo in [15:0], hi in [31:16])
__device__ __forceinline__ unsigned int f2bf2(float lo, float hi) {
#if __has_builtin(__builtin_amdgcn_cvt_pk_bf16_f32)
  auto p = __builtin_amdgcn_cvt_pk_bf16_f32(lo, hi);
  unsigned int r;
  __builtin_memcpy(&r, &p, 4);
  return r;
#else
  unsigned int r;
  asm("v_fma_mixlo_bf16 %0, %1, 1.0, 0" : "=v"(r) : "v"(lo));
  asm("v_fma_mixhi_bf16 %0, %1, 1.0, 0" : "+v"(r) : "v"(hi));
  return r;
#endif
}

__device__ __forceinline__ float gelu_exact(float x) {
  return 0.5f * x * (1.0f + erff(x * 0.70710678118654752f));
}

__device__ __forceinline__ void cvt16(const F4* __restrict__ p, U4* q) {
  F4 f0 = p[0], f1 = p[1], f2 = p[2], f3 = p[3];
  q[0] = U4{ f2bf2(f0.x, f0.y), f2bf2(f0.z, f0.w), f2bf2(f1.x, f1.y), f2bf2(f1.z, f1.w) };
  q[1] = U4{ f2bf2(f2.x, f2.y), f2bf2(f2.z, f2.w), f2bf2(f3.x, f3.y), f2bf2(f3.z, f3.w) };
}

// ---------------------------------------------------------------- router ----
__global__ void moe_router_kernel(const float* __restrict__ xf,
                                  const float* __restrict__ wg,
                                  int* __restrict__ topk_e,
                                  float* __restrict__ topk_p) {
  const int wid  = threadIdx.x >> 5;
  const int lane = threadIdx.x & 31;
  const int n = blockIdx.x * 8 + wid;
  if (n >= NTOK) return;

  float s[NEXPERT];
#pragma unroll
  for (int i = 0; i < NEXPERT; ++i) s[i] = 0.f;

  const float* xr = xf + (size_t)n * DMODEL;
  for (int d = lane; d < DMODEL; d += 32) {
    float xv = xr[d];
    const F4* wr = (const F4*)(wg + (size_t)d * NEXPERT);
    F4 w0 = wr[0], w1 = wr[1];
    s[0] += xv * w0.x; s[1] += xv * w0.y; s[2] += xv * w0.z; s[3] += xv * w0.w;
    s[4] += xv * w1.x; s[5] += xv * w1.y; s[6] += xv * w1.z; s[7] += xv * w1.w;
  }
#pragma unroll
  for (int i = 0; i < NEXPERT; ++i) {
    s[i] += __shfl_xor(s[i], 16, 32);
    s[i] += __shfl_xor(s[i],  8, 32);
    s[i] += __shfl_xor(s[i],  4, 32);
    s[i] += __shfl_xor(s[i],  2, 32);
    s[i] += __shfl_xor(s[i],  1, 32);
  }
  if (lane == 0) {
    int i1 = 0; float v1 = s[0];
#pragma unroll
    for (int i = 1; i < NEXPERT; ++i) if (s[i] > v1) { v1 = s[i]; i1 = i; }
    int i2 = -1; float v2 = -3.4e38f;
#pragma unroll
    for (int i = 0; i < NEXPERT; ++i) if (i != i1 && s[i] > v2) { v2 = s[i]; i2 = i; }
    float e2 = expf(v2 - v1);                 // stable 2-way softmax
    float inv = 1.f / (1.f + e2);
    topk_e[n*2+0] = i1;       topk_e[n*2+1] = i2;
    topk_p[n*2+0] = inv;      topk_p[n*2+1] = e2 * inv;
  }
}

// ------------------------------------------------------- capacity ranking ---
__global__ void moe_assign_kernel(const int* __restrict__ topk_e,
                                  const float* __restrict__ topk_p,
                                  int* __restrict__ slot_tok,
                                  float* __restrict__ slot_w) {
  const int e = threadIdx.x;
  if (e >= NEXPERT) return;
  int* st = slot_tok + e * CAP;
  float* sw = slot_w + e * CAP;
  for (int c = 0; c < CAP; ++c) { st[c] = -1; sw[c] = 0.f; }
  int cnt = 0;
  for (int k = 0; k < 2; ++k)                 // k-major order == reference cumsum
    for (int n = 0; n < NTOK; ++n)
      if (topk_e[n*2+k] == e) {
        if (cnt < CAP) { st[cnt] = n; sw[cnt] = topk_p[n*2+k]; }
        ++cnt;                                // rank advances even when dropped
      }
}

// ------------------------------------------------ GEMM1: gather->fc->gelu ---
// h[c,f] = gelu( sum_d x[tok(c),d] * cfc_e[d,f] ),  tile 128x128, K=DMODEL
__global__ __launch_bounds__(256) void moe_gemm1_kernel(
    const float* __restrict__ xf,
    const float* __restrict__ cfc_e,
    const int* __restrict__ slot_tok_e,
    unsigned short* __restrict__ h) {
  __shared__ __align__(16) unsigned short sA[2 * TBUF];   // [m][k] pitch 40
  __shared__ __align__(16) unsigned short sB[2 * TBUF];   // [n][k] pitch 40

  const int t = threadIdx.x;
  const int mblk = blockIdx.y * 128;
  const int nblk = blockIdx.x * 128;

  // A staging: thread -> one m-row, 16 k-halves
  const int rowA = t >> 1;              // 0..127
  const int kha  = (t & 1) * 16;        // 0 | 16
  const int tokA = slot_tok_e[mblk + rowA];
  const float* aSrc = xf + (size_t)(tokA < 0 ? 0 : tokA) * DMODEL;

  // B staging: thread -> k-pair (k2,k2+1) x 8 n-rows, packed b32 stores
  const int k2  = (t & 15) * 2;         // 0..30
  const int nB0 = (t >> 4) * 8;         // 0..120

  const int wid  = t >> 5, lane = t & 31;
  const int half = lane >> 4, lrow = lane & 15;
  const int wm = wid >> 2, wn = wid & 3;

  // empty slots: zero their LDS rows once (both buffers); they never restage
  if (tokA < 0) {
    U4 z{0u,0u,0u,0u};
    U4* d0 = (U4*)&sA[rowA * APITCH + kha];
    U4* d1 = (U4*)&sA[TBUF + rowA * APITCH + kha];
    d0[0] = z; d0[1] = z; d1[0] = z; d1[1] = z;
  }

  f32x8 acc[4][2];
#pragma unroll
  for (int i = 0; i < 4; ++i)
#pragma unroll
    for (int j = 0; j < 2; ++j)
#pragma unroll
      for (int q = 0; q < 8; ++q) acc[i][j][q] = 0.f;

  auto stage = [&](int bb, int k0) {
    if (tokA >= 0) {
      U4 qa[2];
      cvt16((const F4*)(aSrc + k0 + kha), qa);
      U4* dst = (U4*)&sA[bb * TBUF + rowA * APITCH + kha];
      dst[0] = qa[0]; dst[1] = qa[1];
    }
    const float* r0 = cfc_e + (size_t)(k0 + k2) * DFF + nblk + nB0;
    const F4* p0 = (const F4*)r0;
    const F4* p1 = (const F4*)(r0 + DFF);
    F4 a0 = p0[0], a1 = p0[1], b0 = p1[0], b1 = p1[1];
    unsigned int pr[8] = {
      f2bf2(a0.x, b0.x), f2bf2(a0.y, b0.y), f2bf2(a0.z, b0.z), f2bf2(a0.w, b0.w),
      f2bf2(a1.x, b1.x), f2bf2(a1.y, b1.y), f2bf2(a1.z, b1.z), f2bf2(a1.w, b1.w) };
    unsigned int* sb32 = (unsigned int*)&sB[bb * TBUF];
#pragma unroll
    for (int i = 0; i < 8; ++i)
      sb32[(nB0 + i) * (APITCH / 2) + (k2 >> 1)] = pr[i];
  };

  auto compute = [&](int bb) {
    const unsigned short* pa = &sA[bb * TBUF];
    const unsigned short* pb = &sB[bb * TBUF];
    Frag fa[4], fb[2];
#pragma unroll
    for (int fm = 0; fm < 4; ++fm) {
      int aoff = (wm * 64 + fm * 16 + lrow) * APITCH + half * 8;
      fa[fm].q[0] = *(const U4*)&pa[aoff];
      fa[fm].q[1] = *(const U4*)&pa[aoff + 16];
    }
#pragma unroll
    for (int fn = 0; fn < 2; ++fn) {
      int boff = (wn * 32 + fn * 16 + lrow) * APITCH + half * 8;
      fb[fn].q[0] = *(const U4*)&pb[boff];
      fb[fn].q[1] = *(const U4*)&pb[boff + 16];
    }
#pragma unroll
    for (int fm = 0; fm < 4; ++fm)
#pragma unroll
      for (int fn = 0; fn < 2; ++fn)
        acc[fm][fn] = __builtin_amdgcn_wmma_f32_16x16x32_bf16(
            false, fa[fm].v, false, fb[fn].v, (short)0, acc[fm][fn], false, false);
  };

  stage(0, 0);
  __syncthreads();
  int buf = 0;
  for (int k0 = 32; k0 < DMODEL; k0 += 32) {
    stage(buf ^ 1, k0);     // prefetch next tile into other buffer
    compute(buf);           // WMMA overlaps load latency
    __syncthreads();
    buf ^= 1;
  }
  compute(buf);

#pragma unroll
  for (int fm = 0; fm < 4; ++fm)
#pragma unroll
    for (int fn = 0; fn < 2; ++fn) {
      int col = nblk + wn * 32 + fn * 16 + lrow;
#pragma unroll
      for (int j = 0; j < 8; ++j) {
        int row = mblk + wm * 64 + fm * 16 + half * 8 + j;
        h[(size_t)row * DFF + col] = f2bf(gelu_exact(acc[fm][fn][j]));
      }
    }
}

// ------------------------------------- GEMM2: proj + weighted combine -------
// out_acc[tok(c),d] += w(c) * sum_f h[c,f] * cpj_e[f,d]
// tile 64(M)x128(N), grid 8x32 = 256 WGs for full-machine occupancy.
__global__ __launch_bounds__(256) void moe_gemm2_kernel(
    const unsigned short* __restrict__ h,
    const float* __restrict__ cpj_e,
    const int* __restrict__ slot_tok_e,
    const float* __restrict__ slot_w_e,
    float* __restrict__ out_acc) {
  __shared__ __align__(16) unsigned short sA[2 * T64BUF];  // 64 rows
  __shared__ __align__(16) unsigned short sB[2 * TBUF];    // 128 rows

  const int t = threadIdx.x;
  const int mblk = blockIdx.y * 64;
  const int nblk = blockIdx.x * 128;

  // A staging: one async b128 per thread (64 rows x 32 halves)
  const int rowA = t >> 2;              // 0..63
  const int kqa  = (t & 3) * 8;         // 0,8,16,24 halves (16B aligned)
  // B staging: k-pair x 8 n-rows, packed b32 stores
  const int k2  = (t & 15) * 2;
  const int nB0 = (t >> 4) * 8;

  const int wid  = t >> 5, lane = t & 31;
  const int half = lane >> 4, lrow = lane & 15;
  const int wm = wid >> 2, wn = wid & 3;   // waves: 2(M) x 4(N), wave tile 32x32

  f32x8 acc[2][2];
#pragma unroll
  for (int i = 0; i < 2; ++i)
#pragma unroll
    for (int j = 0; j < 2; ++j)
#pragma unroll
      for (int q = 0; q < 8; ++q) acc[i][j][q] = 0.f;

  auto stage = [&](int bb, int k0) {
    // A tile: straight bf16 copy -> CDNA5 async global->LDS (ASYNCcnt path)
    {
      const unsigned short* g0 = h + (size_t)(mblk + rowA) * DFF + k0 + kqa;
      unsigned int lds0 =
          (unsigned int)(unsigned long long)(const void*)&sA[bb * T64BUF + rowA * APITCH + kqa];
      asm volatile("global_load_async_to_lds_b128 %0, %1, off"
                   :: "v"(lds0), "v"(g0) : "memory");
    }
    // B tile: f32 -> bf16 pair-packed transpose, conflict-free b32 stores
    const float* r0 = cpj_e + (size_t)(k0 + k2) * DMODEL + nblk + nB0;
    const F4* p0 = (const F4*)r0;
    const F4* p1 = (const F4*)(r0 + DMODEL);
    F4 a0 = p0[0], a1 = p0[1], b0 = p1[0], b1 = p1[1];
    unsigned int pr[8] = {
      f2bf2(a0.x, b0.x), f2bf2(a0.y, b0.y), f2bf2(a0.z, b0.z), f2bf2(a0.w, b0.w),
      f2bf2(a1.x, b1.x), f2bf2(a1.y, b1.y), f2bf2(a1.z, b1.z), f2bf2(a1.w, b1.w) };
    unsigned int* sb32 = (unsigned int*)&sB[bb * TBUF];
#pragma unroll
    for (int i = 0; i < 8; ++i)
      sb32[(nB0 + i) * (APITCH / 2) + (k2 >> 1)] = pr[i];
  };

  auto compute = [&](int bb) {
    const unsigned short* pa = &sA[bb * T64BUF];
    const unsigned short* pb = &sB[bb * TBUF];
    Frag fa[2], fb[2];
#pragma unroll
    for (int fm = 0; fm < 2; ++fm) {
      int aoff = (wm * 32 + fm * 16 + lrow) * APITCH + half * 8;
      fa[fm].q[0] = *(const U4*)&pa[aoff];
      fa[fm].q[1] = *(const U4*)&pa[aoff + 16];
    }
#pragma unroll
    for (int fn = 0; fn < 2; ++fn) {
      int boff = (wn * 32 + fn * 16 + lrow) * APITCH + half * 8;
      fb[fn].q[0] = *(const U4*)&pb[boff];
      fb[fn].q[1] = *(const U4*)&pb[boff + 16];
    }
#pragma unroll
    for (int fm = 0; fm < 2; ++fm)
#pragma unroll
      for (int fn = 0; fn < 2; ++fn)
        acc[fm][fn] = __builtin_amdgcn_wmma_f32_16x16x32_bf16(
            false, fa[fm].v, false, fb[fn].v, (short)0, acc[fm][fn], false, false);
  };

  stage(0, 0);
  asm volatile("s_wait_asynccnt 0x0" ::: "memory");
  __syncthreads();
  int buf = 0;
  for (int k0 = 32; k0 < DFF; k0 += 32) {
    stage(buf ^ 1, k0);
    compute(buf);
    asm volatile("s_wait_asynccnt 0x0" ::: "memory");
    __syncthreads();
    buf ^= 1;
  }
  compute(buf);

  // fused combine: one writer per (token,col) within this expert; experts
  // are serialized on the stream -> no atomics needed.
#pragma unroll
  for (int fm = 0; fm < 2; ++fm) {
    int rbase = mblk + wm * 32 + fm * 16 + half * 8;
#pragma unroll
    for (int j = 0; j < 8; ++j) {
      int slot = rbase + j;
      int tok = slot_tok_e[slot];
      if (tok < 0) continue;
      float w = slot_w_e[slot];
#pragma unroll
      for (int fn = 0; fn < 2; ++fn) {
        int col = nblk + wn * 32 + fn * 16 + lrow;
        float* p = out_acc + (size_t)tok * DMODEL + col;
        *p += w * acc[fm][fn][j];
      }
    }
  }
}

// ------------------------------------------------------------- LayerNorm ----
__global__ void moe_ln_kernel(const float* __restrict__ out_acc,
                              const float* __restrict__ ln_w,
                              const float* __restrict__ ln_b,
                              float* __restrict__ out) {
  const int wid = threadIdx.x >> 5, lane = threadIdx.x & 31;
  const int n = blockIdx.x * 8 + wid;
  if (n >= NTOK) return;
  const float* r = out_acc + (size_t)n * DMODEL;
  float v[32];
  float sum = 0.f, sq = 0.f;
#pragma unroll
  for (int i = 0; i < 32; ++i) {
    float x = r[lane + i * 32];
    v[i] = x; sum += x; sq += x * x;
  }
  sum += __shfl_xor(sum, 16, 32); sq += __shfl_xor(sq, 16, 32);
  sum += __shfl_xor(sum,  8, 32); sq += __shfl_xor(sq,  8, 32);
  sum += __shfl_xor(sum,  4, 32); sq += __shfl_xor(sq,  4, 32);
  sum += __shfl_xor(sum,  2, 32); sq += __shfl_xor(sq,  2, 32);
  sum += __shfl_xor(sum,  1, 32); sq += __shfl_xor(sq,  1, 32);
  const float mean = sum * (1.f / DMODEL);
  const float var  = sq * (1.f / DMODEL) - mean * mean;
  const float rstd = rsqrtf(var + LN_EPS);
  float* o = out + (size_t)n * DMODEL;
#pragma unroll
  for (int i = 0; i < 32; ++i) {
    int d = lane + i * 32;
    o[d] = (v[i] - mean) * rstd * ln_w[d] + ln_b[d];
  }
}

// ----------------------------------------------------------------------------
extern "C" void kernel_launch(void* const* d_in, const int* in_sizes, int n_in,
                              void* d_out, int out_size, void* d_ws, size_t ws_size,
                              hipStream_t stream) {
  (void)in_sizes; (void)n_in; (void)out_size; (void)ws_size;
  const float* x   = (const float*)d_in[0];   // [4,1024,1024]
  const float* wg  = (const float*)d_in[1];   // [1024,8]
  const float* cfc = (const float*)d_in[2];   // [8,1024,4096]
  const float* cpj = (const float*)d_in[3];   // [8,4096,1024]
  const float* lnw = (const float*)d_in[4];
  const float* lnb = (const float*)d_in[5];
  float* out = (float*)d_out;

  char* ws = (char*)d_ws;
  size_t off = 0;
  auto alloc = [&](size_t bytes) -> void* {
    off = (off + 255) & ~(size_t)255;
    void* p = ws + off;
    off += bytes;
    return p;
  };
  int*   topk_e   = (int*)  alloc((size_t)NTOK * 2 * sizeof(int));
  float* topk_p   = (float*)alloc((size_t)NTOK * 2 * sizeof(float));
  int*   slot_tok = (int*)  alloc((size_t)NEXPERT * CAP * sizeof(int));
  float* slot_w   = (float*)alloc((size_t)NEXPERT * CAP * sizeof(float));
  float* out_acc  = (float*)alloc((size_t)NTOK * DMODEL * sizeof(float));
  unsigned short* hbuf = (unsigned short*)alloc((size_t)CAP * DFF * sizeof(unsigned short));

  moe_router_kernel<<<NTOK / 8, 256, 0, stream>>>(x, wg, topk_e, topk_p);
  moe_assign_kernel<<<1, 32, 0, stream>>>(topk_e, topk_p, slot_tok, slot_w);
  hipMemsetAsync(out_acc, 0, (size_t)NTOK * DMODEL * sizeof(float), stream);

  for (int e = 0; e < NEXPERT; ++e) {
    moe_gemm1_kernel<<<dim3(DFF / 128, CAP / 128), 256, 0, stream>>>(
        x, cfc + (size_t)e * DMODEL * DFF, slot_tok + e * CAP, hbuf);
    moe_gemm2_kernel<<<dim3(DMODEL / 128, CAP / 64), 256, 0, stream>>>(
        hbuf, cpj + (size_t)e * DFF * DMODEL, slot_tok + e * CAP,
        slot_w + e * CAP, out_acc);
  }

  moe_ln_kernel<<<NTOK / 8, 256, 0, stream>>>(out_acc, lnw, lnb, out);
}
